// TritonSNNClassifier_33595234189782
// MI455X (gfx1250) — compile-verified, matrix operand order checked by reading, not run on hardware
//
#include <hip/hip_runtime.h>

typedef __attribute__((ext_vector_type(16))) _Float16 v16h;
typedef __attribute__((ext_vector_type(8)))  _Float16 v8h;
typedef __attribute__((ext_vector_type(8)))  float    v8f;

#define SNN_B    512
#define SNN_DIN  2048
#define SNN_DH   4096
#define SNN_DOUT 1024
#define SNN_T    16
#define SNN_BETA 0.95f
#define SNN_THR  1.0f

#define LDSP 40   // halves per staged row: 32 data + 8 pad (80B -> 20-bank stride)

// ---------------------------------------------------------------------------
// gfx1250 async copy: 16B per lane, global -> LDS, tracked by ASYNCcnt.
// LDS operand = wave-relative LDS byte offset (low 32 bits of generic addr).
// ---------------------------------------------------------------------------
__device__ __forceinline__ void async_b128_raw(unsigned loff, unsigned long long ga) {
  asm volatile("global_load_async_to_lds_b128 %0, %1, off"
               :: "v"(loff), "v"(ga) : "memory");
}
__device__ __forceinline__ void wait_async0() {
  asm volatile("s_wait_asynccnt 0" ::: "memory");
}

// ---------------------------------------------------------------------------
// 16-bit A/B fragment of V_WMMA_*_16X16X32 from an LDS tile (row-major,
// LDSP-half pitch).  Lanes 0-15: row M=lane, K = {kh+0..7, kh+16..23},
// kh = 8*(lane>>4).  Two ds_load_b128 per lane.
// ---------------------------------------------------------------------------
__device__ __forceinline__ v16h lds_frag(const _Float16* tile0, int lane) {
  const int r  = lane & 15;
  const int kh = (lane >> 4) << 3;
  const _Float16* p = tile0 + r * LDSP + kh;
  union { v16h v; v8h h[2]; } u;
  u.h[0] = *(const v8h*)(p);
  u.h[1] = *(const v8h*)(p + 16);
  return u.v;
}

__device__ __forceinline__ v8f wmma_f16(v16h a, v16h b, v8f c) {
  return __builtin_amdgcn_wmma_f32_16x16x32_f16(
      false, a, false, b, (short)0, c, false, false);
}

// ---------------------------------------------------------------------------
// Precision-split converters (grid-stride)
// ---------------------------------------------------------------------------
__global__ void snn_split_hl_kernel(const float* __restrict__ src,
                                    _Float16* __restrict__ hi,
                                    _Float16* __restrict__ lo, int n) {
  for (int i = blockIdx.x * blockDim.x + threadIdx.x; i < n;
       i += gridDim.x * blockDim.x) {
    float f = src[i];
    _Float16 h = (_Float16)f;
    hi[i] = h;
    lo[i] = (_Float16)(f - (float)h);
  }
}

__global__ void snn_cvt_f16_kernel(const float* __restrict__ src,
                                   _Float16* __restrict__ dst, int n) {
  for (int i = blockIdx.x * blockDim.x + threadIdx.x; i < n;
       i += gridDim.x * blockDim.x) {
    dst[i] = (_Float16)src[i];
  }
}

// ---------------------------------------------------------------------------
// GEMM1 + fused LIF:  cnt = LIF16( x @ W1^T + b1 ) as f16 spike counts.
// Split-f16 (xh*Wh + xh*Wl + xl*Wh ~ f32).  Block tile 64x128, K-chunk 32,
// double-buffered async LDS staging.  8 waves = 2(m) x 4(n); wave = 32x32 out.
// Grid: (512/64)*(4096/128) = 256 blocks.
// ---------------------------------------------------------------------------
__global__ void __launch_bounds__(256)
snn_fc1_lif_kernel(const _Float16* __restrict__ xh, const _Float16* __restrict__ xl,
                   const _Float16* __restrict__ w1h, const _Float16* __restrict__ w1l,
                   const float* __restrict__ b1, _Float16* __restrict__ cnt) {
  // rows [0,64)=Ah  [64,128)=Al  [128,256)=Bh  [256,384)=Bl
  __shared__ _Float16 sm[2][384 * LDSP];

  const int tid  = threadIdx.x;
  const int lane = tid & 31;
  const int wid  = tid >> 5;
  const int wm   = wid & 1;
  const int wn   = wid >> 1;
  const int m0   = (blockIdx.x & 7) << 6;
  const int n0   = (blockIdx.x >> 3) << 7;

  // Hoisted copy descriptors: 1536 16B chunks / 256 threads = 6 per thread.
  unsigned long long ga[6];
  unsigned           lofs[6];
#pragma unroll
  for (int i = 0; i < 6; ++i) {
    const int c   = tid + (i << 8);
    const int col = c & 3;
    const _Float16* g;
    int lrow;
    if (c < 512) {                          // Ah then Al, 64 rows each
      const int a   = c >> 8;
      const int row = (c >> 2) & 63;
      g = (a ? xl : xh) + (size_t)(m0 + row) * SNN_DIN + (size_t)(col * 8);
      lrow = (a << 6) + row;
    } else {                                // Bh then Bl, 128 rows each
      const int c2  = c - 512;
      const int b   = c2 >> 9;
      const int row = (c2 >> 2) & 127;
      g = (b ? w1l : w1h) + (size_t)(n0 + row) * SNN_DIN + (size_t)(col * 8);
      lrow = 128 + (b << 7) + row;
    }
    ga[i]   = (unsigned long long)g;
    lofs[i] = (unsigned)(unsigned long long)&sm[0][lrow * LDSP + col * 8];
  }
  const unsigned bufstride = (unsigned)(384 * LDSP * sizeof(_Float16));

  auto stage = [&](int buf) {               // issues one K-chunk, advances K by 32
    const unsigned bo = buf ? bufstride : 0u;
#pragma unroll
    for (int i = 0; i < 6; ++i) {
      async_b128_raw(lofs[i] + bo, ga[i]);
      ga[i] += 64;                          // 32 halves
    }
  };

  v8f acc[2][2];
#pragma unroll
  for (int f = 0; f < 2; ++f)
#pragma unroll
    for (int t = 0; t < 2; ++t) acc[f][t] = (v8f){};

  stage(0);
  for (int ks = 0; ks < SNN_DIN / 32; ++ks) {
    const int cur = ks & 1;
    wait_async0();                          // this wave's stage `ks` landed
    __syncthreads();                        // all landed; prior reads of other buf done
    if (ks + 1 < SNN_DIN / 32) stage(cur ^ 1);

    const _Float16* buf = sm[cur];
    v16h ahf[2], alf[2], bhf[2], blf[2];
#pragma unroll
    for (int f = 0; f < 2; ++f) {
      const int ar = ((wm << 1) + f) << 4;
      ahf[f] = lds_frag(buf + ar * LDSP, lane);
      alf[f] = lds_frag(buf + (64 + ar) * LDSP, lane);
    }
#pragma unroll
    for (int t = 0; t < 2; ++t) {
      const int br = ((wn << 1) + t) << 4;
      bhf[t] = lds_frag(buf + (128 + br) * LDSP, lane);
      blf[t] = lds_frag(buf + (256 + br) * LDSP, lane);
    }
#pragma unroll
    for (int f = 0; f < 2; ++f)
#pragma unroll
      for (int t = 0; t < 2; ++t) {
        acc[f][t] = wmma_f16(ahf[f], bhf[t], acc[f][t]);
        acc[f][t] = wmma_f16(ahf[f], blf[t], acc[f][t]);
        acc[f][t] = wmma_f16(alf[f], bhf[t], acc[f][t]);
      }
  }

  // Closed-form LIF: mem_t = cur * sum_{i<t} beta^i  (mem0=0, no reset in ref),
  // so spike_t  <=>  cur > THR / c_t.  Constant-folded thresholds.
  float thr[SNN_T];
  {
    float csum = 0.0f, p = 1.0f;
#pragma unroll
    for (int s = 0; s < SNN_T; ++s) { csum += p; p *= SNN_BETA; thr[s] = SNN_THR / csum; }
  }

  const int n  = lane & 15;
  const int mb = (lane >> 4) << 3;
#pragma unroll
  for (int f = 0; f < 2; ++f)
#pragma unroll
    for (int t = 0; t < 2; ++t)
#pragma unroll
      for (int v = 0; v < 8; ++v) {
        const int row = m0 + (((wm << 1) + f) << 4) + mb + v;
        const int col = n0 + (((wn << 1) + t) << 4) + n;
        const float curv = acc[f][t][v] + b1[col];
        float c = 0.0f;
#pragma unroll
        for (int s = 0; s < SNN_T; ++s) c += (curv > thr[s]) ? 1.0f : 0.0f;
        cnt[(size_t)row * SNN_DH + col] = (_Float16)c;   // exact int in [0,16]
      }
}

// ---------------------------------------------------------------------------
// GEMM2: out = (cnt @ W2^T)/16 + b2.  One GEMM replaces 16 per-step GEMMs;
// counts 0..16 are exact in f16.  Same 64x128 async-LDS tiling.
// Grid: (512/64)*(1024/128) = 64 blocks.
// ---------------------------------------------------------------------------
__global__ void __launch_bounds__(256)
snn_fc2_kernel(const _Float16* __restrict__ cnt, const _Float16* __restrict__ w2h,
               const float* __restrict__ b2, float* __restrict__ out) {
  // rows [0,64)=A   [64,192)=B
  __shared__ _Float16 sm[2][192 * LDSP];

  const int tid  = threadIdx.x;
  const int lane = tid & 31;
  const int wid  = tid >> 5;
  const int wm   = wid & 1;
  const int wn   = wid >> 1;
  const int m0   = (blockIdx.x & 7) << 6;
  const int n0   = (blockIdx.x >> 3) << 7;

  unsigned long long ga[3];
  unsigned           lofs[3];
#pragma unroll
  for (int i = 0; i < 3; ++i) {
    const int c   = tid + (i << 8);
    const int col = c & 3;
    const _Float16* g;
    int lrow;
    if (c < 256) {                          // A: 64 rows
      const int row = c >> 2;
      g = cnt + (size_t)(m0 + row) * SNN_DH + (size_t)(col * 8);
      lrow = row;
    } else {                                // B: 128 rows
      const int c2  = c - 256;
      const int row = c2 >> 2;
      g = w2h + (size_t)(n0 + row) * SNN_DH + (size_t)(col * 8);
      lrow = 64 + row;
    }
    ga[i]   = (unsigned long long)g;
    lofs[i] = (unsigned)(unsigned long long)&sm[0][lrow * LDSP + col * 8];
  }
  const unsigned bufstride = (unsigned)(192 * LDSP * sizeof(_Float16));

  auto stage = [&](int buf) {
    const unsigned bo = buf ? bufstride : 0u;
#pragma unroll
    for (int i = 0; i < 3; ++i) {
      async_b128_raw(lofs[i] + bo, ga[i]);
      ga[i] += 64;
    }
  };

  v8f acc[2][2];
#pragma unroll
  for (int f = 0; f < 2; ++f)
#pragma unroll
    for (int t = 0; t < 2; ++t) acc[f][t] = (v8f){};

  stage(0);
  for (int ks = 0; ks < SNN_DH / 32; ++ks) {
    const int cur = ks & 1;
    wait_async0();
    __syncthreads();
    if (ks + 1 < SNN_DH / 32) stage(cur ^ 1);

    const _Float16* buf = sm[cur];
    v16h a[2], b[2];
#pragma unroll
    for (int f = 0; f < 2; ++f)
      a[f] = lds_frag(buf + ((((wm << 1) + f) << 4)) * LDSP, lane);
#pragma unroll
    for (int t = 0; t < 2; ++t)
      b[t] = lds_frag(buf + (64 + (((wn << 1) + t) << 4)) * LDSP, lane);
#pragma unroll
    for (int f = 0; f < 2; ++f)
#pragma unroll
      for (int t = 0; t < 2; ++t)
        acc[f][t] = wmma_f16(a[f], b[t], acc[f][t]);
  }

  const int n  = lane & 15;
  const int mb = (lane >> 4) << 3;
#pragma unroll
  for (int f = 0; f < 2; ++f)
#pragma unroll
    for (int t = 0; t < 2; ++t)
#pragma unroll
      for (int v = 0; v < 8; ++v) {
        const int row = m0 + (((wm << 1) + f) << 4) + mb + v;
        const int col = n0 + (((wn << 1) + t) << 4) + n;
        out[(size_t)row * SNN_DOUT + col] =
            acc[f][t][v] * (1.0f / (float)SNN_T) + b2[col];
      }
}

// ---------------------------------------------------------------------------
// Launch
// ---------------------------------------------------------------------------
extern "C" void kernel_launch(void* const* d_in, const int* in_sizes, int n_in,
                              void* d_out, int out_size, void* d_ws, size_t ws_size,
                              hipStream_t stream) {
  (void)in_sizes; (void)n_in; (void)out_size; (void)ws_size;

  const float* x  = (const float*)d_in[0];   // [512, 2048]
  const float* W1 = (const float*)d_in[1];   // [4096, 2048]
  const float* b1 = (const float*)d_in[2];   // [4096]
  const float* W2 = (const float*)d_in[3];   // [1024, 4096]
  const float* b2 = (const float*)d_in[4];   // [1024]
  float* out = (float*)d_out;                // [512, 1024]

  char* ws = (char*)d_ws;
  _Float16* xh  = (_Float16*)ws; ws += (size_t)SNN_B    * SNN_DIN * 2;
  _Float16* xl  = (_Float16*)ws; ws += (size_t)SNN_B    * SNN_DIN * 2;
  _Float16* w1h = (_Float16*)ws; ws += (size_t)SNN_DH   * SNN_DIN * 2;
  _Float16* w1l = (_Float16*)ws; ws += (size_t)SNN_DH   * SNN_DIN * 2;
  _Float16* w2h = (_Float16*)ws; ws += (size_t)SNN_DOUT * SNN_DH  * 2;
  _Float16* cnt = (_Float16*)ws; ws += (size_t)SNN_B    * SNN_DH  * 2;

  snn_split_hl_kernel<<<2048, 256, 0, stream>>>(x,  xh,  xl,  SNN_B * SNN_DIN);
  snn_split_hl_kernel<<<4096, 256, 0, stream>>>(W1, w1h, w1l, SNN_DH * SNN_DIN);
  snn_cvt_f16_kernel <<<4096, 256, 0, stream>>>(W2, w2h, SNN_DOUT * SNN_DH);

  snn_fc1_lif_kernel<<<256, 256, 0, stream>>>(xh, xl, w1h, w1l, b1, cnt);
  snn_fc2_kernel<<<64, 256, 0, stream>>>(cnt, w2h, b2, out);
}
